// ExtractionNet_38525856645669
// MI455X (gfx1250) — matrix operand order repaired and checked by reading, not compiled, not run on hardware
//
#include <hip/hip_runtime.h>

typedef __bf16 bf16;
typedef __attribute__((ext_vector_type(16))) __bf16 bf16x16;
typedef __attribute__((ext_vector_type(8)))  float  f32x8;

enum { Bz = 8192, Dz = 1024, H1z = 2048, H2z = 1024, H3z = 512,
       NSz = 8, NTz = 4, Ez = 20, TINz = 2048, TU1z = 1024, TU2z = 512 };

// ---------------------------------------------------------------------------
// CDNA5 async global->LDS DMA (ASYNCcnt-tracked), 16B per lane.
#if defined(__HIP_DEVICE_COMPILE__)
__device__ __forceinline__ void async_load_b128(const bf16* gptr, bf16* lptr) {
  auto l3 = (__attribute__((address_space(3))) bf16*)lptr;   // generic -> LDS
  asm volatile("global_load_async_to_lds_b128 %0, %1, off"
               :: "v"(l3), "v"(gptr) : "memory");
}
__device__ __forceinline__ void wait_async0() {
  asm volatile("s_wait_asynccnt 0" ::: "memory");
}
#else
__device__ __forceinline__ void async_load_b128(const bf16*, bf16*) {}
__device__ __forceinline__ void wait_async0() {}
#endif

// ---------------------------------------------------------------------------
// fp32 -> bf16 elementwise convert
__global__ void convert_bf16(const float* __restrict__ src, bf16* __restrict__ dst, int n) {
  int i = blockIdx.x * blockDim.x + threadIdx.x;
  if (i < n) dst[i] = (bf16)src[i];
}

// ---------------------------------------------------------------------------
// fp32 W [K,N] (row-major) -> bf16 Wt [N,K] (row-major), 64x64 LDS tile.
// grid = (N/64, K/64, Z); per-z matrices are contiguous (K*N elements).
__global__ __launch_bounds__(256) void transpose_convert(
    const float* __restrict__ W, bf16* __restrict__ Wt, int K, int N) {
  __shared__ __align__(16) bf16 tile[64][72];  // +8 pad breaks bank conflicts
  size_t zoff = (size_t)blockIdx.z * (size_t)K * (size_t)N;
  const float* Wz = W + zoff;
  bf16* Wtz = Wt + zoff;
  int n0 = blockIdx.x * 64, k0 = blockIdx.y * 64;
  int t = threadIdx.x;
  int r  = t >> 2;          // 0..63
  int cg = (t & 3) * 16;    // 0,16,32,48
  const float* src = Wz + (size_t)(k0 + r) * N + n0 + cg;
#pragma unroll
  for (int j = 0; j < 16; ++j) tile[r][cg + j] = (bf16)src[j];
  __syncthreads();
  bf16 tmp[16];
#pragma unroll
  for (int j = 0; j < 16; ++j) tmp[j] = tile[cg + j][r];
  bf16* dst = Wtz + (size_t)(n0 + r) * K + k0 + cg;   // 32B aligned
  *(uint4*)dst       = *(uint4*)&tmp[0];
  *(uint4*)(dst + 8) = *(uint4*)&tmp[8];
}

// ---------------------------------------------------------------------------
// C = lrelu(A @ W + bias):  A [M,K] bf16 row-major, Wt [N,K] bf16 row-major
// (W pre-transposed), C [M,N] bf16.  M%128==0, N%128==0, K%32==0.
// Block: 256 thr = 8 waves (4 M x 2 N); wave tile 32x64 = 2x4 WMMA 16x16 tiles.
// Double-buffered LDS fed by async global->LDS DMA; one barrier per K-step.
__global__ __launch_bounds__(256) void gemm_bf16_lrelu(
    const bf16* __restrict__ A, const bf16* __restrict__ Wt,
    const float* __restrict__ bias, bf16* __restrict__ C,
    int M, int N, int K) {
  __shared__ __align__(16) bf16 lA[2][128 * 40];  // row stride 40 bf16 = 80B
  __shared__ __align__(16) bf16 lB[2][128 * 40];

  const int tid  = threadIdx.x;
  const int lane = tid & 31;
  const int wid  = tid >> 5;
  const int waveM = wid >> 1;        // 0..3
  const int waveN = wid & 1;         // 0..1
  const int l16 = lane & 15;
  const int hi  = lane >> 4;         // 0/1

  const int m0 = blockIdx.y * 128;
  const int n0 = blockIdx.x * 128;

  // staging map: each thread DMAs one 16B chunk per tile
  const int sr = tid >> 1;           // 0..127
  const int sh = (tid & 1) * 16;     // 0 / 16 (elements)
  const int soff = sr * 40 + sh;     // LDS element offset (16B aligned)
  const bf16* gA = A  + (size_t)(m0 + sr) * K + sh;
  const bf16* gB = Wt + (size_t)(n0 + sr) * K + sh;

  f32x8 acc[2][4];
#pragma unroll
  for (int mt = 0; mt < 2; ++mt)
#pragma unroll
    for (int nt = 0; nt < 4; ++nt) acc[mt][nt] = (f32x8){0,0,0,0,0,0,0,0};

  // prologue: DMA first K-tile into buffer 0
  async_load_b128(gA, &lA[0][soff]);
  async_load_b128(gB, &lB[0][soff]);
  wait_async0();
  __syncthreads();

  const int nsteps = K >> 5;
  int cur = 0;
  for (int i = 0; i < nsteps; ++i) {
    if (i + 1 < nsteps) {  // prefetch next K-tile into the other buffer
      async_load_b128(gA + (size_t)(i + 1) * 32, &lA[1 - cur][soff]);
      async_load_b128(gB + (size_t)(i + 1) * 32, &lB[1 - cur][soff]);
    }

    // A fragments: 16x32, lane(hi,m): elems 0-7 <-> k=8*hi+0..7, 8-15 <-> 16+8*hi+0..7
    bf16x16 af[2];
#pragma unroll
    for (int mt = 0; mt < 2; ++mt) {
      const bf16* p = &lA[cur][(waveM * 32 + mt * 16 + l16) * 40 + 8 * hi];
      ((uint4*)&af[mt])[0] = *(const uint4*)p;
      ((uint4*)&af[mt])[1] = *(const uint4*)(p + 16);
    }
    // B fragments: 32x16, lane(hi,n): elems 0-7 <-> k=8*hi+0..7, 8-15 <-> 16+8*hi+0..7
    bf16x16 bfr[4];
#pragma unroll
    for (int nt = 0; nt < 4; ++nt) {
      const bf16* p = &lB[cur][(waveN * 64 + nt * 16 + l16) * 40 + 8 * hi];
      ((uint4*)&bfr[nt])[0] = *(const uint4*)p;
      ((uint4*)&bfr[nt])[1] = *(const uint4*)(p + 16);
    }
#pragma unroll
    for (int mt = 0; mt < 2; ++mt)
#pragma unroll
      for (int nt = 0; nt < 4; ++nt)
        acc[mt][nt] = __builtin_amdgcn_wmma_f32_16x16x32_bf16(
            false, af[mt], false, bfr[nt], (short)0, acc[mt][nt], false, false);

    if (i + 1 < nsteps) {
      wait_async0();     // next buffer landed in LDS
      __syncthreads();   // all waves done reading cur & done DMAing next
      cur ^= 1;
    }
  }

  // epilogue: D layout -> VGPR r holds (M = r + 8*hi, N = l16) per 16x16 tile
#pragma unroll
  for (int nt = 0; nt < 4; ++nt) {
    const int col = n0 + waveN * 64 + nt * 16 + l16;
    const float bn = bias[col];
#pragma unroll
    for (int mt = 0; mt < 2; ++mt) {
      const int rbase = m0 + waveM * 32 + mt * 16 + 8 * hi;
#pragma unroll
      for (int r = 0; r < 8; ++r) {
        float v = acc[mt][nt][r] + bn;
        v = (v >= 0.f) ? v : 0.1f * v;
        C[(size_t)(rbase + r) * N + col] = (bf16)v;
      }
    }
  }
}

// ---------------------------------------------------------------------------
// gate logits: gates[B][56] = [sh(20) | sa(12) | ra(12) | ea(12)]
__global__ void gate_logits(const float* __restrict__ x,
                            const float* __restrict__ Wsh, const float* __restrict__ bsh,
                            const float* __restrict__ Wsa, const float* __restrict__ bsa,
                            const float* __restrict__ Wra, const float* __restrict__ bra,
                            const float* __restrict__ Wea, const float* __restrict__ bea,
                            float* __restrict__ gates) {
  int idx = blockIdx.x * blockDim.x + threadIdx.x;
  if (idx >= Bz * 56) return;
  int b = idx / 56, c = idx % 56;
  const float* W; const float* bi; int g, gc;
  if (c < 20)      { W = Wsh; bi = bsh; g = c;      gc = 20; }
  else if (c < 32) { W = Wsa; bi = bsa; g = c - 20; gc = 12; }
  else if (c < 44) { W = Wra; bi = bra; g = c - 32; gc = 12; }
  else             { W = Wea; bi = bea; g = c - 44; gc = 12; }
  float acc = bi[g];
  const float* xr = x + (size_t)b * Dz;
  for (int k = 0; k < Dz; ++k) acc += xr[k] * W[k * gc + g];
  gates[idx] = acc;
}

__global__ void gate_softmax(float* __restrict__ gates) {
  int idx = blockIdx.x * blockDim.x + threadIdx.x;
  if (idx >= Bz * 4) return;
  int b = idx >> 2, s = idx & 3;
  const int off[4] = {0, 20, 32, 44};
  int n = (s == 0) ? 20 : 12;
  float* p = gates + (size_t)b * 56 + off[s];
  float m = p[0];
  for (int i = 1; i < n; ++i) m = fmaxf(m, p[i]);
  float sum = 0.f;
  for (int i = 0; i < n; ++i) { float e = __expf(p[i] - m); p[i] = e; sum += e; }
  float inv = 1.f / sum;
  for (int i = 0; i < n; ++i) p[i] *= inv;
}

// ---------------------------------------------------------------------------
// gated mixes -> tower_in [B, 2048] bf16 (segments: sh | sa | ra | ea)
__global__ void combine_experts(const bf16* __restrict__ eo,  // [E][B][512]
                                const float* __restrict__ gates,
                                bf16* __restrict__ tin) {
  int idx = blockIdx.x * blockDim.x + threadIdx.x;
  if (idx >= Bz * TINz) return;
  int b = idx / TINz, c = idx % TINz;
  int s = c / H3z, h = c % H3z;
  const float* g = gates + (size_t)b * 56;
  float acc = 0.f;
  if (s == 0) {
    for (int e = 0; e < Ez; ++e)
      acc += g[e] * (float)eo[((size_t)e * Bz + b) * H3z + h];
  } else {
    int gbase = 20 + (s - 1) * 12;
    for (int i = 0; i < 12; ++i) {
      int e = (i < NTz) ? (NSz + (s - 1) * NTz + i) : (i - NTz);
      acc += g[gbase + i] * (float)eo[((size_t)e * Bz + b) * H3z + h];
    }
  }
  tin[idx] = (bf16)acc;
}

// ---------------------------------------------------------------------------
// final small dense: out[b,j] = t2[b,:] @ Wd[:,j] + bd[j]
__global__ void dense_out(const bf16* __restrict__ t2, const float* __restrict__ Wd,
                          const float* __restrict__ bd, float* __restrict__ out, int ncol) {
  int idx = blockIdx.x * blockDim.x + threadIdx.x;
  if (idx >= Bz * ncol) return;
  int b = idx / ncol, j = idx % ncol;
  float acc = bd[j];
  const bf16* tr = t2 + (size_t)b * TU2z;
  for (int k = 0; k < TU2z; ++k) acc += (float)tr[k] * Wd[k * ncol + j];
  out[idx] = acc;
}

// ---------------------------------------------------------------------------
extern "C" void kernel_launch(void* const* d_in, const int* in_sizes, int n_in,
                              void* d_out, int out_size, void* d_ws, size_t ws_size,
                              hipStream_t stream) {
  const float* x    = (const float*)d_in[0];
  const float* W1   = (const float*)d_in[1];
  const float* b1   = (const float*)d_in[2];
  const float* W2   = (const float*)d_in[3];
  const float* b2   = (const float*)d_in[4];
  const float* W3   = (const float*)d_in[5];
  const float* b3   = (const float*)d_in[6];
  const float* Wgsh = (const float*)d_in[7];
  const float* bgsh = (const float*)d_in[8];
  const float* Wgsa = (const float*)d_in[9];
  const float* bgsa = (const float*)d_in[10];
  const float* Wgra = (const float*)d_in[11];
  const float* bgra = (const float*)d_in[12];
  const float* Wgea = (const float*)d_in[13];
  const float* bgea = (const float*)d_in[14];
  const float* Wt1  = (const float*)d_in[15];
  const float* bt1  = (const float*)d_in[16];
  const float* Wt2  = (const float*)d_in[17];
  const float* bt2  = (const float*)d_in[18];
  const float* Wd0  = (const float*)d_in[19];
  const float* bd0  = (const float*)d_in[20];
  const float* Wd1  = (const float*)d_in[21];
  const float* bd1  = (const float*)d_in[22];
  (void)in_sizes; (void)n_in; (void)out_size; (void)ws_size;

  char* w = (char*)d_ws;
  auto alloc = [&](size_t bytes) -> void* {
    void* p = (void*)w; w += (bytes + 255) & ~(size_t)255; return p;
  };
  bf16* xb   = (bf16*)alloc((size_t)Bz * Dz * 2);
  bf16* W1t  = (bf16*)alloc((size_t)Ez * Dz * H1z * 2);
  bf16* W2t  = (bf16*)alloc((size_t)Ez * H1z * H2z * 2);
  bf16* W3t  = (bf16*)alloc((size_t)Ez * H2z * H3z * 2);
  bf16* Wt1t = (bf16*)alloc((size_t)2 * TINz * TU1z * 2);
  bf16* Wt2t = (bf16*)alloc((size_t)2 * TU1z * TU2z * 2);
  bf16* h1   = (bf16*)alloc((size_t)Bz * H1z * 2);
  bf16* h2   = (bf16*)alloc((size_t)Bz * H2z * 2);
  bf16* eo   = (bf16*)alloc((size_t)Ez * Bz * H3z * 2);
  float* gts = (float*)alloc((size_t)Bz * 56 * 4);
  bf16* tin  = (bf16*)alloc((size_t)Bz * TINz * 2);
  bf16* t1   = (bf16*)alloc((size_t)Bz * TU1z * 2);
  bf16* t2   = (bf16*)alloc((size_t)Bz * TU2z * 2);

  // --- conversions ---
  {
    int n = Bz * Dz;
    convert_bf16<<<(n + 255) / 256, 256, 0, stream>>>(x, xb, n);
  }
  transpose_convert<<<dim3(H1z / 64, Dz / 64, Ez),  256, 0, stream>>>(W1,  W1t,  Dz,  H1z);
  transpose_convert<<<dim3(H2z / 64, H1z / 64, Ez), 256, 0, stream>>>(W2,  W2t,  H1z, H2z);
  transpose_convert<<<dim3(H3z / 64, H2z / 64, Ez), 256, 0, stream>>>(W3,  W3t,  H2z, H3z);
  transpose_convert<<<dim3(TU1z / 64, TINz / 64, 2), 256, 0, stream>>>(Wt1, Wt1t, TINz, TU1z);
  transpose_convert<<<dim3(TU2z / 64, TU1z / 64, 2), 256, 0, stream>>>(Wt2, Wt2t, TU1z, TU2z);

  // --- gates (independent of experts) ---
  {
    int n = Bz * 56;
    gate_logits<<<(n + 255) / 256, 256, 0, stream>>>(x, Wgsh, bgsh, Wgsa, bgsa,
                                                     Wgra, bgra, Wgea, bgea, gts);
    int m = Bz * 4;
    gate_softmax<<<(m + 255) / 256, 256, 0, stream>>>(gts);
  }

  // --- 20 experts: 3 WMMA GEMMs each ---
  for (int e = 0; e < Ez; ++e) {
    gemm_bf16_lrelu<<<dim3(H1z / 128, Bz / 128), 256, 0, stream>>>(
        xb, W1t + (size_t)e * Dz * H1z, b1 + (size_t)e * H1z, h1, Bz, H1z, Dz);
    gemm_bf16_lrelu<<<dim3(H2z / 128, Bz / 128), 256, 0, stream>>>(
        h1, W2t + (size_t)e * H1z * H2z, b2 + (size_t)e * H2z, h2, Bz, H2z, H1z);
    gemm_bf16_lrelu<<<dim3(H3z / 128, Bz / 128), 256, 0, stream>>>(
        h2, W3t + (size_t)e * H2z * H3z, b3 + (size_t)e * H3z,
        eo + (size_t)e * Bz * H3z, Bz, H3z, H2z);
  }

  // --- gated combine -> tower input ---
  {
    int n = Bz * TINz;
    combine_experts<<<(n + 255) / 256, 256, 0, stream>>>(eo, gts, tin);
  }

  // --- towers + output denses ---
  float* out0 = (float*)d_out;          // [B,6]
  float* out1 = out0 + (size_t)Bz * 6;  // [B,4]
  for (int k = 0; k < 2; ++k) {
    gemm_bf16_lrelu<<<dim3(TU1z / 128, Bz / 128), 256, 0, stream>>>(
        tin, Wt1t + (size_t)k * TINz * TU1z, bt1 + (size_t)k * TU1z, t1, Bz, TU1z, TINz);
    gemm_bf16_lrelu<<<dim3(TU2z / 128, Bz / 128), 256, 0, stream>>>(
        t1, Wt2t + (size_t)k * TU1z * TU2z, bt2 + (size_t)k * TU2z, t2, Bz, TU2z, TU1z);
    if (k == 0) {
      int n = Bz * 6;
      dense_out<<<(n + 255) / 256, 256, 0, stream>>>(t2, Wd0, bd0, out0, 6);
    } else {
      int n = Bz * 4;
      dense_out<<<(n + 255) / 256, 256, 0, stream>>>(t2, Wd1, bd1, out1, 4);
    }
  }
}
